// MA_DNC_78572131713101
// MI455X (gfx1250) — matrix operand order repaired and checked by reading, not compiled, not run on hardware
//
#include <hip/hip_runtime.h>
#include <cstddef>

// ---------------------------------------------------------------------------
// Model dims
// ---------------------------------------------------------------------------
#define DD   512
#define NN   16
#define SS   2056            // 4*D + 8
#define CLIPV 20.0f
#define EPSF  1e-6f
#define DELTAF 5e-6f
#define TOK_START 145

// DNC state slot layout (floats)
#define SLOT_CHX   0
#define SLOT_MEM   512
#define SLOT_LINK  8704
#define SLOT_PREC  8960
#define SLOT_RW    8976
#define SLOT_WW    8992
#define SLOT_USAGE 9008
#define SLOT_LR    9024
#define SLOT_SIZE  9600      // padded (multiple of 16 floats)

typedef __bf16 v16bf __attribute__((ext_vector_type(16)));
typedef float  v8f   __attribute__((ext_vector_type(8)));
typedef float  flt4  __attribute__((ext_vector_type(4)));

__device__ __forceinline__ float sigmf_(float x) { return 1.0f / (1.0f + expf(-x)); }
__device__ __forceinline__ float softplusf_(float x) { return (x > 20.0f) ? x : log1pf(expf(x)); }

// A fragment for one 32-wide K chunk, contiguous k order (same as loadB32):
//   lane row = m0+(lane&15); K = kpos + 16*hi + e
__device__ __forceinline__ v16bf loadA32(const float* __restrict__ Wrow,
                                         int kpos, int hi, bool ok)
{
    v16bf a;
    if (ok) {
        const flt4* p0 = (const flt4*)(Wrow + kpos + hi * 16);
        const flt4* p1 = (const flt4*)(Wrow + kpos + 8 + hi * 16);
        flt4 a0 = p0[0], a1 = p0[1], b0 = p1[0], b1 = p1[1];
        #pragma unroll
        for (int e = 0; e < 4; ++e) {
            a[e]      = (__bf16)a0[e];
            a[4 + e]  = (__bf16)a1[e];
            a[8 + e]  = (__bf16)b0[e];
            a[12 + e] = (__bf16)b1[e];
        }
    } else {
        #pragma unroll
        for (int e = 0; e < 16; ++e) a[e] = (__bf16)0.0f;
    }
    return a;
}

// B fragment: every column = x[kglob .. kglob+31]; this lane holds K = 16*hi + e.
__device__ __forceinline__ v16bf loadB32(const float* __restrict__ sp,
                                         const float* __restrict__ addv,
                                         int kglob, int hi, int actIn)
{
    const flt4* p = (const flt4*)(sp + hi * 16);
    flt4 v[4] = { p[0], p[1], p[2], p[3] };
    if (actIn) {
        #pragma unroll
        for (int j = 0; j < 4; ++j)
            #pragma unroll
            for (int e = 0; e < 4; ++e) v[j][e] = fmaxf(v[j][e], 0.0f);
    }
    if (addv) {
        const flt4* q = (const flt4*)(addv + kglob + hi * 16);
        #pragma unroll
        for (int j = 0; j < 4; ++j) {
            flt4 w = q[j];
            #pragma unroll
            for (int e = 0; e < 4; ++e) v[j][e] += w[e];
        }
    }
    v16bf b;
    #pragma unroll
    for (int j = 0; j < 4; ++j)
        #pragma unroll
        for (int e = 0; e < 4; ++e) b[4 * j + e] = (__bf16)v[j][e];
    return b;
}

// ---------------------------------------------------------------------------
// GEMV via bf16 WMMA 16x16x32, K unrolled x4 with independent accumulators.
// y[M] = actOut( W[MxK] @ f(x) + bias ),  f(x) = [relu?](concat(x0..x3)) + addv
// One wave computes a 16-row tile. Weights streamed fp32 (L2-resident on
// MI455X: total model ~130MB < 192MB L2), converted to bf16 in-register.
// ---------------------------------------------------------------------------
__global__ __launch_bounds__(256)
void gemv_wmma_kernel(const float* __restrict__ W, int M, int K,
                      const float* x0, int l0, const float* x1, int l1,
                      const float* x2, int l2, const float* x3, int l3,
                      const float* __restrict__ addv, const float* __restrict__ bias,
                      float* __restrict__ y, int actIn, int actOut)
{
    const int lane = threadIdx.x & 31;
    const int wave = blockIdx.x * (blockDim.x >> 5) + (threadIdx.x >> 5);
    const int m0 = wave * 16;
    if (m0 >= M) return;                    // wave-uniform: EXEC all-ones for WMMA

    const int  hi    = lane >> 4;           // 0: low half-wave, 1: high half-wave
    const int  row   = m0 + (lane & 15);
    const bool rowOK = row < M;
    const float* Wrow = W + (size_t)row * (size_t)K;

    v8f acc0 = {}, acc1 = {}, acc2 = {}, acc3 = {};

    int kk = 0;
    // Main loop: 128 K per iteration, 4 independent WMMA accumulators.
    // Segments are multiples of 512, so a 128-aligned chunk never crosses one.
    for (; kk + 128 <= K; kk += 128) {
        int rem = kk;
        const float* sp;
        if (rem < l0) sp = x0 + rem;
        else { rem -= l0;
            if (rem < l1) sp = x1 + rem;
            else { rem -= l1;
                if (rem < l2) sp = x2 + rem;
                else          sp = x3 + (rem - l2);
            }
        }
        if (rowOK && kk + 384 <= K)
            __builtin_prefetch(Wrow + kk + 256, 0, 0);   // global_prefetch

        v16bf a0 = loadA32(Wrow, kk,      hi, rowOK);
        v16bf b0 = loadB32(sp,       addv, kk,      hi, actIn);
        v16bf a1 = loadA32(Wrow, kk + 32, hi, rowOK);
        v16bf b1 = loadB32(sp + 32,  addv, kk + 32, hi, actIn);
        v16bf a2 = loadA32(Wrow, kk + 64, hi, rowOK);
        v16bf b2 = loadB32(sp + 64,  addv, kk + 64, hi, actIn);
        v16bf a3 = loadA32(Wrow, kk + 96, hi, rowOK);
        v16bf b3 = loadB32(sp + 96,  addv, kk + 96, hi, actIn);

        acc0 = __builtin_amdgcn_wmma_f32_16x16x32_bf16(false, a0, false, b0, (short)0, acc0, false, false);
        acc1 = __builtin_amdgcn_wmma_f32_16x16x32_bf16(false, a1, false, b1, (short)0, acc1, false, false);
        acc2 = __builtin_amdgcn_wmma_f32_16x16x32_bf16(false, a2, false, b2, (short)0, acc2, false, false);
        acc3 = __builtin_amdgcn_wmma_f32_16x16x32_bf16(false, a3, false, b3, (short)0, acc3, false, false);
    }
    // Generic tail (unused for this model's K values, all multiples of 128).
    for (; kk < K; kk += 32) {
        int rem = kk;
        const float* sp;
        if (rem < l0) sp = x0 + rem;
        else { rem -= l0;
            if (rem < l1) sp = x1 + rem;
            else { rem -= l1;
                if (rem < l2) sp = x2 + rem;
                else          sp = x3 + (rem - l2);
            }
        }
        v16bf a = loadA32(Wrow, kk, hi, rowOK);
        v16bf b = loadB32(sp, addv, kk, hi, actIn);
        acc0 = __builtin_amdgcn_wmma_f32_16x16x32_bf16(false, a, false, b, (short)0, acc0, false, false);
    }

    v8f acc;
    #pragma unroll
    for (int r = 0; r < 8; ++r) acc[r] = (acc0[r] + acc1[r]) + (acc2[r] + acc3[r]);

    // D layout: VGPR r, lane L -> M = r + 8*(L>=16), N = L&15. Column-0 lanes write.
    if ((lane & 15) == 0) {
        #pragma unroll
        for (int r = 0; r < 8; ++r) {
            int m = m0 + r + 8 * hi;
            if (m < M) {
                float v = acc[r];
                if (bias)   v += bias[m];
                if (actOut) v = fmaxf(v, 0.0f);
                y[m] = v;
            }
        }
    }
}

// ---------------------------------------------------------------------------
// GRU gate combine: h' = (1-z)*n + z*h ; optional clipped copy.
// ---------------------------------------------------------------------------
__global__ void gru_combine_kernel(const float* __restrict__ gi,
                                   const float* __restrict__ gh,
                                   const float* __restrict__ hin,
                                   float* __restrict__ hout,
                                   float* __restrict__ hclip,
                                   int H, float clipv)
{
    int i = blockIdx.x * blockDim.x + threadIdx.x;
    if (i >= H) return;
    float r = sigmf_(gi[i] + gh[i]);
    float z = sigmf_(gi[H + i] + gh[H + i]);
    float n = tanhf(gi[2 * H + i] + r * gh[2 * H + i]);
    float h = (1.0f - z) * n + z * hin[i];
    hout[i] = h;
    if (hclip) hclip[i] = fminf(fmaxf(h, -clipv), clipv);
}

// ---------------------------------------------------------------------------
// Embedding row gather (token index read on-device). Launch <<<1, 512>>>.
// ---------------------------------------------------------------------------
__global__ void gather_row_kernel(const float* __restrict__ emb,
                                  const int* __restrict__ idx, int off, int cidx,
                                  float* __restrict__ dst)
{
    int t = idx ? idx[off] : cidx;
    dst[threadIdx.x] = emb[(size_t)t * DD + threadIdx.x];
}

// ---------------------------------------------------------------------------
// Full DNC memory step, single block of 512 threads. Reads state Sin, writes
// Sout (Sin != Sout). Memory matrix resident in LDS; LDS-based reductions.
// ---------------------------------------------------------------------------
__global__ __launch_bounds__(512)
void memory_step_kernel(const float* __restrict__ xi,
                        const float* __restrict__ Sin,
                        float* __restrict__ Sout)
{
    const int tid  = threadIdx.x;
    const int lane = tid & 31;
    const int wid  = tid >> 5;

    __shared__ float s_rkey[DD], s_wkey[DD], s_erase[DD], s_wvec[DD];
    __shared__ float s_mem[NN * DD];
    __shared__ float s_link[NN * NN];
    __shared__ float s_part[512];
    __shared__ float s_scal[8]; // rd_str, wr_str, fg, ag, wg, rm0, rm1, rm2
    __shared__ float s_dot[NN], s_nrm[NN];
    __shared__ float s_knrm;
    __shared__ float s_usage[NN], s_ww[NN], s_wcw[NN], s_rcw[NN];
    __shared__ float s_rwold[NN], s_rwnew[NN], s_alloc[NN];

    // ---- parse interface vector ----
    s_rkey[tid]  = tanhf(xi[tid]);
    s_wkey[tid]  = tanhf(xi[513 + tid]);
    s_erase[tid] = sigmf_(xi[1026 + tid]);
    s_wvec[tid]  = tanhf(xi[1538 + tid]);
    for (int i = 0; i < NN; ++i) s_mem[i * DD + tid] = Sin[SLOT_MEM + i * DD + tid];
    if (tid == 0) {
        s_scal[0] = softplusf_(xi[512]);
        s_scal[1] = softplusf_(xi[1025]);
        s_scal[2] = sigmf_(xi[2050]);
        s_scal[3] = sigmf_(xi[2051]);
        s_scal[4] = sigmf_(xi[2052]);
        float a = xi[2053], b = xi[2054], c = xi[2055];
        float m = fmaxf(a, fmaxf(b, c));
        float ea = expf(a - m), eb = expf(b - m), ec = expf(c - m);
        float dn = ea + eb + ec;
        s_scal[5] = ea / dn; s_scal[6] = eb / dn; s_scal[7] = ec / dn;
    }
    __syncthreads();

    // ---- usage update (uses old ww, old rw) ----
    if (tid < NN) {
        float u  = Sin[SLOT_USAGE + tid];
        float w0 = Sin[SLOT_WW + tid];
        float r0 = Sin[SLOT_RW + tid];
        u = u + (1.0f - u) * w0;
        u = u * (1.0f - s_scal[2] * r0);
        s_usage[tid] = u;
        s_rwold[tid] = r0;
    }

    // ---- cosine(old mem, write_key): wave wid handles row wid ----
    {
        float d = 0, n2 = 0, k2 = 0;
        for (int e = 0; e < DD / 32; ++e) {
            float m = s_mem[wid * DD + lane + 32 * e];
            float k = s_wkey[lane + 32 * e];
            d += m * k; n2 += m * m; k2 += k * k;
        }
        s_part[tid] = d; __syncthreads();
        if (lane == 0) { float s = 0; for (int l = 0; l < 32; ++l) s += s_part[wid * 32 + l]; s_dot[wid] = s; }
        __syncthreads();
        s_part[tid] = n2; __syncthreads();
        if (lane == 0) { float s = 0; for (int l = 0; l < 32; ++l) s += s_part[wid * 32 + l]; s_nrm[wid] = s; }
        __syncthreads();
        s_part[tid] = k2; __syncthreads();
        if (tid == 0) { float s = 0; for (int l = 0; l < 32; ++l) s += s_part[l]; s_knrm = s; }
        __syncthreads();
    }

    // ---- wcw softmax + allocation (N=16: serial on one lane) ----
    if (tid == 0) {
        float kn = sqrtf(s_knrm) + EPSF;
        float z[NN]; float mx = -1e30f;
        for (int i = 0; i < NN; ++i) {
            float sim = s_dot[i] / ((sqrtf(s_nrm[i]) + EPSF) * kn);
            z[i] = s_scal[1] * sim; mx = fmaxf(mx, z[i]);
        }
        float sum = 0;
        for (int i = 0; i < NN; ++i) { z[i] = expf(z[i] - mx); sum += z[i]; }
        for (int i = 0; i < NN; ++i) s_wcw[i] = z[i] / sum;
        // allocation: ascending sort of u, exclusive cumprod
        float u[NN]; int ord[NN];
        for (int i = 0; i < NN; ++i) { u[i] = DELTAF + (1.0f - DELTAF) * s_usage[i]; ord[i] = i; }
        for (int i = 1; i < NN; ++i) {
            float uv = u[i]; int ov = ord[i]; int j = i - 1;
            while (j >= 0 && u[j] > uv) { u[j + 1] = u[j]; ord[j + 1] = ord[j]; --j; }
            u[j + 1] = uv; ord[j + 1] = ov;
        }
        float ex = 1.0f;
        for (int i = 0; i < NN; ++i) { s_alloc[ord[i]] = (1.0f - u[i]) * ex; ex *= u[i]; }
    }
    __syncthreads();

    // ---- write weighting ----
    if (tid < NN) {
        float w = s_scal[4] * (s_scal[3] * s_alloc[tid] + (1.0f - s_scal[3]) * s_wcw[tid]);
        s_ww[tid] = w;
        Sout[SLOT_WW + tid]    = w;
        Sout[SLOT_USAGE + tid] = s_usage[tid];
    }
    __syncthreads();

    // ---- erase / write memory (thread owns column tid) ----
    {
        float er = s_erase[tid], wv = s_wvec[tid];
        for (int i = 0; i < NN; ++i) {
            float m = s_mem[i * DD + tid];
            float w = s_ww[i];
            m = m * (1.0f - w * er) + w * wv;
            s_mem[i * DD + tid] = m;
            Sout[SLOT_MEM + i * DD + tid] = m;
        }
    }
    // ---- link + precedence (old link/prec, new ww) ----
    if (tid < NN * NN) {
        int i = tid >> 4, j = tid & 15;
        float ln = (1.0f - s_ww[i] - s_ww[j]) * Sin[SLOT_LINK + tid]
                 + s_ww[i] * Sin[SLOT_PREC + j];
        if (i == j) ln = 0.0f;
        s_link[tid] = ln;
        Sout[SLOT_LINK + tid] = ln;
    }
    if (tid < NN) {
        float wsum = 0; for (int i = 0; i < NN; ++i) wsum += s_ww[i];
        Sout[SLOT_PREC + tid] = (1.0f - wsum) * Sin[SLOT_PREC + tid] + s_ww[tid];
    }
    __syncthreads();

    // ---- cosine(new mem, read_key) ----
    {
        float d = 0, n2 = 0, k2 = 0;
        for (int e = 0; e < DD / 32; ++e) {
            float m = s_mem[wid * DD + lane + 32 * e];
            float k = s_rkey[lane + 32 * e];
            d += m * k; n2 += m * m; k2 += k * k;
        }
        s_part[tid] = d; __syncthreads();
        if (lane == 0) { float s = 0; for (int l = 0; l < 32; ++l) s += s_part[wid * 32 + l]; s_dot[wid] = s; }
        __syncthreads();
        s_part[tid] = n2; __syncthreads();
        if (lane == 0) { float s = 0; for (int l = 0; l < 32; ++l) s += s_part[wid * 32 + l]; s_nrm[wid] = s; }
        __syncthreads();
        s_part[tid] = k2; __syncthreads();
        if (tid == 0) { float s = 0; for (int l = 0; l < 32; ++l) s += s_part[l]; s_knrm = s; }
        __syncthreads();
    }
    if (tid == 0) {
        float kn = sqrtf(s_knrm) + EPSF;
        float z[NN]; float mx = -1e30f;
        for (int i = 0; i < NN; ++i) {
            float sim = s_dot[i] / ((sqrtf(s_nrm[i]) + EPSF) * kn);
            z[i] = s_scal[0] * sim; mx = fmaxf(mx, z[i]);
        }
        float sum = 0;
        for (int i = 0; i < NN; ++i) { z[i] = expf(z[i] - mx); sum += z[i]; }
        for (int i = 0; i < NN; ++i) s_rcw[i] = z[i] / sum;
    }
    __syncthreads();

    // ---- directional reads + new read weighting ----
    if (tid < NN) {
        float fwd = 0, bwd = 0;
        for (int j = 0; j < NN; ++j) {
            fwd += s_link[tid * NN + j] * s_rwold[j];
            bwd += s_rwold[j] * s_link[j * NN + tid];
        }
        float rn = s_scal[5] * bwd + s_scal[6] * fwd + s_scal[7] * s_rcw[tid];
        s_rwnew[tid] = rn;
        Sout[SLOT_RW + tid] = rn;
    }
    __syncthreads();

    // ---- read vector ----
    {
        float rv = 0;
        for (int i = 0; i < NN; ++i) rv += s_rwnew[i] * s_mem[i * DD + tid];
        Sout[SLOT_LR + tid] = rv;
    }
}

// ---------------------------------------------------------------------------
// Host orchestration
// ---------------------------------------------------------------------------
static void launch_gemv(hipStream_t st, const float* W, int M, int K,
                        const float* x0, int l0, const float* x1, int l1,
                        const float* x2, int l2, const float* x3, int l3,
                        const float* addv, const float* bias, float* y,
                        int actIn, int actOut)
{
    int tiles = (M + 15) / 16;
    int wavesPerBlk = 8;                   // 256 threads = 8 wave32
    int blocks = (tiles + wavesPerBlk - 1) / wavesPerBlk;
    gemv_wmma_kernel<<<blocks, wavesPerBlk * 32, 0, st>>>(
        W, M, K, x0, l0, x1, l1, x2, l2, x3, l3, addv, bias, y, actIn, actOut);
}

struct EncP {
    const float *W_ih, *W_hh, *b_ih, *b_hh, *W_int, *b_int, *W_out, *b_out;
};

extern "C" void kernel_launch(void* const* d_in, const int* in_sizes, int n_in,
                              void* d_out, int out_size, void* d_ws, size_t ws_size,
                              hipStream_t stream)
{
    (void)n_in; (void)out_size;

    const float* emb0 = (const float*)d_in[0];
    const float* emb1 = (const float*)d_in[1];
    const float* emb2 = (const float*)d_in[2];

    EncP enc[3];
    for (int e = 0; e < 3; ++e) {
        const int b = 3 + e * 8;
        enc[e].W_ih  = (const float*)d_in[b + 0];
        enc[e].W_hh  = (const float*)d_in[b + 1];
        enc[e].b_ih  = (const float*)d_in[b + 2];
        enc[e].b_hh  = (const float*)d_in[b + 3];
        enc[e].W_int = (const float*)d_in[b + 4];
        enc[e].b_int = (const float*)d_in[b + 5];
        enc[e].W_out = (const float*)d_in[b + 6];
        enc[e].b_out = (const float*)d_in[b + 7];
    }
    const float* dec_Wih = (const float*)d_in[27];
    const float* dec_Whh = (const float*)d_in[28];
    const float* dec_bih = (const float*)d_in[29];
    const float* dec_bhh = (const float*)d_in[30];
    const float* r2o_W   = (const float*)d_in[31];
    const float* r2o_b   = (const float*)d_in[32];
    const float* out_W   = (const float*)d_in[33];
    const float* out_b   = (const float*)d_in[34];
    const int* in0 = (const int*)d_in[35];
    const int* in1 = (const int*)d_in[36];
    const int* in2 = (const int*)d_in[37];
    const int T0 = in_sizes[35], T1 = in_sizes[36], T2 = in_sizes[37];

    // ---- workspace layout (floats) ----
    float* wsf = (float*)d_ws;
    auto slotp = [&](int i) { return wsf + (size_t)i * SLOT_SIZE; };
    // slots: 0,1 enc0 pp | 2,3 enc1 pp | 4,5 enc2 pp | 6 zero | 7,8,9 decode scratch
    size_t off = (size_t)10 * SLOT_SIZE;
    float* xbuf = wsf + off; off += 512;
    float* gi   = wsf + off; off += 4608;
    float* gh   = wsf + off; off += 4608;
    float* xib  = wsf + off; off += 2064;
    float* hc   = wsf + off; off += 512;
    float* hdec = wsf + off; off += 1536;
    float* o1   = wsf + off; off += 512;
    float* o2   = wsf + off; off += 512;
    float* o3   = wsf + off; off += 512;
    float* mid  = wsf + off; off += 1536;
    size_t ws_bytes = off * sizeof(float);
    if (ws_bytes > ws_size) ws_bytes = ws_size;
    hipMemsetAsync(d_ws, 0, ws_bytes, stream);   // zero states + hdec each call

    // one full DNC step: gi/gh GEMVs, GRU combine (+clip), interface GEMV,
    // memory step; optionally the W_out projection (decode only).
    auto dnc_step_launch = [&](const EncP& p, const float* xvec,
                               float* sin, float* sout, float* oOut) {
        launch_gemv(stream, p.W_ih, 3 * DD, 2 * DD,
                    xvec, DD, sin + SLOT_LR, DD, nullptr, 0, nullptr, 0,
                    nullptr, p.b_ih, gi, 0, 0);
        launch_gemv(stream, p.W_hh, 3 * DD, DD,
                    sin + SLOT_CHX, DD, nullptr, 0, nullptr, 0, nullptr, 0,
                    nullptr, p.b_hh, gh, 0, 0);
        gru_combine_kernel<<<(DD + 255) / 256, 256, 0, stream>>>(
            gi, gh, sin + SLOT_CHX, sout + SLOT_CHX, hc, DD, CLIPV);
        launch_gemv(stream, p.W_int, SS, DD,
                    hc, DD, nullptr, 0, nullptr, 0, nullptr, 0,
                    nullptr, p.b_int, xib, 0, 0);
        memory_step_kernel<<<1, 512, 0, stream>>>(xib, sin, sout);
        if (oOut)
            launch_gemv(stream, p.W_out, DD, 2 * DD,
                        hc, DD, sout + SLOT_LR, DD, nullptr, 0, nullptr, 0,
                        nullptr, p.b_out, oOut, 0, 0);
    };

    // ---- encode streams 0 and 1 ----
    for (int e = 0; e < 2; ++e) {
        const int*   ids = e ? in1 : in0;
        const int    T   = e ? T1 : T0;
        const float* em  = e ? emb1 : emb0;
        for (int t = 0; t < T; ++t) {
            gather_row_kernel<<<1, 512, 0, stream>>>(em, ids, t, 0, xbuf);
            dnc_step_launch(enc[e], xbuf,
                            slotp(e * 2 + (t & 1)), slotp(e * 2 + ((t + 1) & 1)),
                            nullptr);
        }
    }
    // ---- encoder 2: single START-token step (provides initial r3 only) ----
    gather_row_kernel<<<1, 512, 0, stream>>>(emb2, nullptr, 0, TOK_START, xbuf);
    dnc_step_launch(enc[2], xbuf, slotp(4), slotp(5), nullptr);

    // ---- decode ----
    float* s1f = slotp(0 + (T0 & 1));        // frozen final enc0 state
    float* s2f = slotp(2 + (T1 & 1));        // frozen final enc1 state
    float* zro = slotp(6);                   // frozen zero state (s3_loop)
    const float* r1 = s1f + SLOT_LR;
    const float* r2 = s2f + SLOT_LR;
    const float* r3 = slotp(5) + SLOT_LR;    // s3_start read vector
    float* oB[3] = { o1, o2, o3 };
    float* fz[3] = { s1f, s2f, zro };
    float* logits = (float*)d_out;

    const int steps = T2 + 1;
    for (int t = 0; t < steps; ++t) {
        if (t == 0)
            gather_row_kernel<<<1, 512, 0, stream>>>(emb2, nullptr, 0, TOK_START, xbuf);
        else
            gather_row_kernel<<<1, 512, 0, stream>>>(emb2, in2, t - 1, 0, xbuf);

        // decoder GRU: 4608x2048 and 4608x1536 GEMVs
        launch_gemv(stream, dec_Wih, 9 * DD, 4 * DD,
                    xbuf, DD, r1, DD, r2, DD, r3, DD,
                    nullptr, dec_bih, gi, 0, 0);
        launch_gemv(stream, dec_Whh, 9 * DD, 3 * DD,
                    hdec, 3 * DD, nullptr, 0, nullptr, 0, nullptr, 0,
                    nullptr, dec_bhh, gh, 0, 0);
        gru_combine_kernel<<<(3 * DD + 255) / 256, 256, 0, stream>>>(
            gi, gh, hdec, hdec, nullptr, 3 * DD, 0.0f);

        // three DNC heads, each restarted from its frozen state
        for (int i = 0; i < 3; ++i)
            dnc_step_launch(enc[i], hdec + i * DD, fz[i], slotp(7 + i), oB[i]);

        r1 = slotp(7) + SLOT_LR;
        r2 = slotp(8) + SLOT_LR;
        r3 = slotp(9) + SLOT_LR;

        // mid = relu(r2o_W @ relu([r1,r2,r3]) + b)
        launch_gemv(stream, r2o_W, 3 * DD, 3 * DD,
                    r1, DD, r2, DD, r3, DD, nullptr, 0,
                    nullptr, r2o_b, mid, 1, 1);
        // logits = out_W @ ([o1,o2,o3] + mid) + out_b
        launch_gemv(stream, out_W, TOK_START + 2, 3 * DD,
                    o1, DD, o2, DD, o3, DD, nullptr, 0,
                    mid, out_b, logits + (size_t)t * (TOK_START + 2), 0, 0);
    }
}
